// DGATEncoderGraph_48747878809864
// MI455X (gfx1250) — compile-verified, matrix operand order checked by reading, not run on hardware
//
#include <hip/hip_runtime.h>
#include <hip/hip_bf16.h>
#include <cstddef>
#include <cstdint>

// ---------------------------------------------------------------------------
// GAT (2-layer) for MI455X / gfx1250.
//  - All GEMM-like contractions: v_wmma_f32_16x16x32_bf16 (wave32, f32 acc).
//  - B operands are bf16, stored K-major (transposed) so each B-fragment
//    VGPR (rows k,k+1 of one column) is a single 32-bit LDS word: no cvt.
//  - B/W tiles staged to LDS with GLOBAL_LOAD_ASYNC_TO_LDS_B128 (ASYNCcnt),
//    shared by 4 waves per workgroup.
//  - adj sweeps use global_prefetch + b128 loads.  No atomics.
// ---------------------------------------------------------------------------

typedef __attribute__((ext_vector_type(16))) __bf16    v16bf;
typedef __attribute__((ext_vector_type(8)))  float     v8f;
typedef __attribute__((ext_vector_type(8)))  unsigned  v8u;

#define N_NODES 4096
#define F_IN    256
#define D1      128
#define H1      4
#define D2      256
#define H2      6
#define FIN2    (D1 * H1)   // 512
#define MASK_VAL (-9.0e15f)

__device__ __forceinline__ v8f wmma_bf16(v16bf a, v16bf b, v8f c) {
    return __builtin_amdgcn_wmma_f32_16x16x32_bf16(
        false, a, false, b, (short)0, c, false, false);
}

// k index for element e of a 16-bit A/B fragment held by this lane
// (ISA 7.12.2: lanes 0-15 K-base 0, lanes 16-31 K-base 8; VGPRs 0-3 cover
//  K[base..base+8), VGPRs 4-7 cover K[base+16..base+24)).
__device__ __forceinline__ int frag_k(int e, int half) {
    return ((e >> 3) << 4) + half * 8 + (e & 7);
}

// Async DMA of 16 bytes global -> LDS (ASYNCcnt).
__device__ __forceinline__ void async_load_b128(unsigned lds_byte, const void* gaddr) {
    asm volatile("global_load_async_to_lds_b128 %0, %1, off"
                 :: "v"(lds_byte), "v"(gaddr) : "memory");
}
__device__ __forceinline__ void wait_async() {
    asm volatile("s_wait_asynccnt 0x0" ::: "memory");
}
__device__ __forceinline__ unsigned lds_addr_of(const void* shared_ptr) {
    return (unsigned)(uintptr_t)shared_ptr;   // addr[31:0] == LDS offset
}

// A fragment from row-major bf16 row pointer (element offset = k base 0):
// e<8 run at +half*8, e>=8 run at +16+half*8; each run = 8 bf16 = 16B.
__device__ __forceinline__ v16bf load_a_frag(const __bf16* row, int half) {
    const unsigned* lo = (const unsigned*)(row + half * 8);
    const unsigned* hi = (const unsigned*)(row + 16 + half * 8);
    v8u u;
#pragma unroll
    for (int q = 0; q < 4; ++q) { u[q] = lo[q]; u[4 + q] = hi[q]; }
    return __builtin_bit_cast(v16bf, u);
}

// B fragment from a K-major LDS tile T[cols][32] (bf16): column c, 8 u32.
__device__ __forceinline__ v16bf load_b_frag(const unsigned* T32, int c, int half) {
    v8u u;
#pragma unroll
    for (int r = 0; r < 8; ++r) {
        const int k = frag_k(2 * r, half);          // even
        u[r] = T32[c * 16 + (k >> 1)];
    }
    return __builtin_bit_cast(v16bf, u);
}

// Load 16 floats (lane's P-fragment source run) into a register array.
__device__ __forceinline__ void load_frag_rows(float* dst, const float* p0) {
    *(float4*)(dst + 0)  = *(const float4*)(p0 + 0);
    *(float4*)(dst + 4)  = *(const float4*)(p0 + 4);
    *(float4*)(dst + 8)  = *(const float4*)(p0 + 16);
    *(float4*)(dst + 12) = *(const float4*)(p0 + 20);
}

// ---------------------------------------------------------------------------
// One-time converts: f32 -> bf16 (row-major), and W[h][K][D] -> Wt[h][D][K].
// ---------------------------------------------------------------------------
__global__ void k_f32_to_bf16(const float* __restrict__ src, __bf16* __restrict__ dst,
                              int n) {
    const int i = blockIdx.x * 256 + threadIdx.x;
    if (i < n) dst[i] = (__bf16)src[i];
}
__global__ void k_w_transpose(const float* __restrict__ W, __bf16* __restrict__ Wt,
                              int K, int D, int n) {
    const int i = blockIdx.x * 256 + threadIdx.x;
    if (i >= n) return;
    const int h = i / (K * D), rem = i % (K * D);
    const int k = rem / D, d = rem % D;
    Wt[(size_t)h * K * D + (size_t)d * K + k] = (__bf16)W[i];
}

// ---------------------------------------------------------------------------
// Projection GEMM: Hout[h] = X @ W[h].  X:[N,K] bf16 row-major,
// Wt:[heads,D,K] bf16 (K-major).  Writes f32 row-major Hout[heads,N,D]
// (for scores) and bf16 K-major HoutT[heads,D,N] (attention B operand).
// grid=(N/64, D/16, heads), block=128 (4 waves).
// ---------------------------------------------------------------------------
template <int K, int D>
__global__ __launch_bounds__(128) void k_proj(const __bf16* __restrict__ X,
                                              const __bf16* __restrict__ Wt,
                                              float* __restrict__ Hout,
                                              __bf16* __restrict__ HoutT, int N) {
    __shared__ __align__(16) __bf16 tileW[16 * 32];    // 1 KB, [col][k]
    const int tid  = threadIdx.x;
    const int wave = tid >> 5, lane = tid & 31;
    const int half = lane >> 4, n16 = lane & 15;
    const int row0 = blockIdx.x * 64 + wave * 16;
    const int col0 = blockIdx.y * 16;
    const int h    = blockIdx.z;
    const __bf16* __restrict__ Wth  = Wt + (size_t)h * K * D;
    const __bf16* __restrict__ xrow = X + (size_t)(row0 + n16) * K;
    const unsigned ldsW = lds_addr_of(tileW);
    const unsigned* T32 = (const unsigned*)tileW;

    v8f acc = {0.f,0.f,0.f,0.f,0.f,0.f,0.f,0.f};
    for (int kt = 0; kt < K; kt += 32) {
        __syncthreads();
        if (tid < 64) {   // stage Wt[col0..col0+16, kt..kt+32): 64 b128 chunks
            const int d = col0 + (tid >> 2), part = (tid & 3) * 8;
            async_load_b128(ldsW + (unsigned)(((tid >> 2) * 32 + part) * 2),
                            Wth + (size_t)d * K + kt + part);
        }
        wait_async();
        __syncthreads();

        const v16bf a = load_a_frag(xrow + kt, half);
        const v16bf b = load_b_frag(T32, n16, half);
        acc = wmma_bf16(a, b, acc);
    }
    // f32 row-major store (for score GEMVs)
    float* __restrict__ out = Hout + (size_t)h * N * D;
#pragma unroll
    for (int r = 0; r < 8; ++r) {
        const int M = r + 8 * half;
        out[(size_t)(row0 + M) * D + col0 + n16] = acc[r];
    }
    // bf16 K-major store: 8 consecutive rows per lane -> one b128
    __bf16 packed[8];
#pragma unroll
    for (int r = 0; r < 8; ++r) packed[r] = (__bf16)acc[r];
    __bf16* __restrict__ outT = HoutT + (size_t)h * (size_t)D * N
                              + (size_t)(col0 + n16) * N + row0 + 8 * half;
    *(float4*)outT = *(float4*)packed;
}

// ---------------------------------------------------------------------------
// Per-head attention scores.  grid=(N/256, heads), block=256.
// ---------------------------------------------------------------------------
__global__ void k_scores(const float* __restrict__ H, const float* __restrict__ A,
                         float* __restrict__ ssrc, float* __restrict__ sdst,
                         int N, int D) {
    const int n = blockIdx.x * 256 + threadIdx.x;
    const int h = blockIdx.y;
    const float* __restrict__ hp = H + ((size_t)h * N + n) * D;
    const float* __restrict__ av = A + (size_t)h * 2 * D;
    float s0 = 0.f, s1 = 0.f;
    for (int d = 0; d < D; d += 4) {
        const float4 v  = *(const float4*)(hp + d);
        const float4 a0 = *(const float4*)(av + d);
        const float4 a1 = *(const float4*)(av + D + d);
        s0 = fmaf(v.x, a0.x, fmaf(v.y, a0.y, fmaf(v.z, a0.z, fmaf(v.w, a0.w, s0))));
        s1 = fmaf(v.x, a1.x, fmaf(v.y, a1.y, fmaf(v.z, a1.z, fmaf(v.w, a1.w, s1))));
    }
    ssrc[(size_t)h * N + n] = s0;
    sdst[(size_t)h * N + n] = s1;
}

// ---------------------------------------------------------------------------
// Row softmax stats (online max/sum).  grid=(N/256, heads), block=256.
// ---------------------------------------------------------------------------
__global__ void k_stats(const float* __restrict__ adj,
                        const float* __restrict__ ssrc, const float* __restrict__ sdst,
                        const float* __restrict__ acv, const float* __restrict__ bcv,
                        float* __restrict__ mOut, float* __restrict__ zOut, int N) {
    const int i = blockIdx.x * 256 + threadIdx.x;
    const int h = blockIdx.y;
    const float si = ssrc[(size_t)h * N + i];
    const float ac = acv[h], bc = bcv[h];
    const float* __restrict__ arow = adj + (size_t)i * N;
    const float* __restrict__ sd = sdst + (size_t)h * N;
    float m = -3.0e38f, z = 0.f;
    for (int j0 = 0; j0 < N; j0 += 64) {
        if (j0 + 512 < N) __builtin_prefetch(arow + j0 + 512, 0, 3);
#pragma unroll 4
        for (int j = j0; j < j0 + 64; j += 4) {
            const float4 a4 = *(const float4*)(arow + j);
            const float4 s4 = *(const float4*)(sd + j);
            const float av[4] = {a4.x, a4.y, a4.z, a4.w};
            const float sv[4] = {s4.x, s4.y, s4.z, s4.w};
#pragma unroll
            for (int q = 0; q < 4; ++q) {
                const float aij = av[q];
                const float e   = si + sv[q];
                const float t   = fmaf(ac, aij, bc);
                const float lk  = (t >= 0.f) ? t : 0.2f * t;
                const float att = (aij > 0.f) ? lk * e : MASK_VAL;
                if (att > m) { z = z * __expf(m - att) + 1.f; m = att; }
                else         { z += __expf(att - m); }
            }
        }
    }
    mOut[(size_t)h * N + i] = m;
    zOut[(size_t)h * N + i] = z;
}

// ---------------------------------------------------------------------------
// Layer-1 attention apply (concat=True): h' = softmax(att) @ H1, ELU,
// write bf16 concat buffer [N, H1*D1] (only consumed as a GEMM A operand).
// HT:[H1][D1][N] bf16 K-major.  grid=(N/64, H1), block=128 (4 waves).
// ---------------------------------------------------------------------------
__global__ __launch_bounds__(128) void k_attn1(
        const float* __restrict__ adj, const __bf16* __restrict__ HT,
        const float* __restrict__ ssrc, const float* __restrict__ sdst,
        const float* __restrict__ mv, const float* __restrict__ zv,
        const float* __restrict__ acv, const float* __restrict__ bcv,
        __bf16* __restrict__ Hcat, int N) {
    __shared__ __align__(16) __bf16 tileB[D1 * 32];    // 8 KB, [col][k]
    const int tid  = threadIdx.x;
    const int wave = tid >> 5, lane = tid & 31;
    const int half = lane >> 4, n16 = lane & 15;
    const int i0   = blockIdx.x * 64 + wave * 16;
    const int h    = blockIdx.y;
    const int irow = i0 + n16;
    const float si = ssrc[(size_t)h * N + irow];
    const float mi = mv[(size_t)h * N + irow];
    const float rz = 1.f / zv[(size_t)h * N + irow];
    const float ac = acv[h], bc = bcv[h];
    const float* __restrict__ arow = adj + (size_t)irow * N;
    const float* __restrict__ sd   = sdst + (size_t)h * N;
    const __bf16* __restrict__ Hpt = HT + (size_t)h * (size_t)D1 * N;
    const unsigned ldsB = lds_addr_of(tileB);
    const unsigned* T32 = (const unsigned*)tileB;

    v8f acc[8];
#pragma unroll
    for (int t = 0; t < 8; ++t) acc[t] = (v8f){0.f,0.f,0.f,0.f,0.f,0.f,0.f,0.f};

    for (int jt = 0; jt < N; jt += 32) {
        __syncthreads();
        {   // stage Hpt[:, jt..jt+32): D1 rows x 64B = 512 b128 chunks
#pragma unroll
            for (int c = 0; c < (D1 * 4) / 128; ++c) {
                const int idx = c * 128 + tid;
                const int d = idx >> 2, part = (idx & 3) * 8;
                async_load_b128(ldsB + (unsigned)idx * 16,
                                Hpt + (size_t)d * N + jt + part);
            }
        }
        wait_async();
        __syncthreads();

        float av[16], sv[16];
        load_frag_rows(av, arow + jt + half * 8);
        load_frag_rows(sv, sd + jt + half * 8);
        v16bf P;
#pragma unroll
        for (int e = 0; e < 16; ++e) {
            const float aij = av[e];
            const float eij = si + sv[e];
            const float t   = fmaf(ac, aij, bc);
            const float lk  = (t >= 0.f) ? t : 0.2f * t;
            const float att = (aij > 0.f) ? lk * eij : MASK_VAL;
            P[e] = (__bf16)(__expf(att - mi) * rz);
        }
#pragma unroll
        for (int t = 0; t < 8; ++t) {
            const v16bf B = load_b_frag(T32, t * 16 + n16, half);
            acc[t] = wmma_bf16(P, B, acc[t]);
        }
    }
    const int stride = H1 * D1;
#pragma unroll
    for (int t = 0; t < 8; ++t) {
#pragma unroll
        for (int r = 0; r < 8; ++r) {
            const int M = r + 8 * half;
            const float v = acc[t][r];
            const float o = (v > 0.f) ? v : (__expf(v) - 1.f);   // ELU
            Hcat[(size_t)(i0 + M) * stride + h * D1 + t * 16 + n16] = (__bf16)o;
        }
    }
}

// ---------------------------------------------------------------------------
// Layer-2 attention apply (concat=False): out = ELU(mean_h softmax@H2_h).
// HT:[H2][D2][N] bf16 K-major.  grid=N/64, block=128 (4 waves).
// ---------------------------------------------------------------------------
__global__ __launch_bounds__(128) void k_attn2(
        const float* __restrict__ adj, const __bf16* __restrict__ HT,
        const float* __restrict__ ssrc, const float* __restrict__ sdst,
        const float* __restrict__ mv, const float* __restrict__ zv,
        const float* __restrict__ acv, const float* __restrict__ bcv,
        float* __restrict__ OutBuf, int N) {
    __shared__ __align__(16) __bf16 tileB[D2 * 32];    // 16 KB, [col][k]
    const int tid  = threadIdx.x;
    const int wave = tid >> 5, lane = tid & 31;
    const int half = lane >> 4, n16 = lane & 15;
    const int i0   = blockIdx.x * 64 + wave * 16;
    const int irow = i0 + n16;
    const float* __restrict__ arow = adj + (size_t)irow * N;
    const unsigned ldsB = lds_addr_of(tileB);
    const unsigned* T32 = (const unsigned*)tileB;

    v8f acc[16];
#pragma unroll
    for (int t = 0; t < 16; ++t) acc[t] = (v8f){0.f,0.f,0.f,0.f,0.f,0.f,0.f,0.f};

    for (int h = 0; h < H2; ++h) {
        const float si = ssrc[(size_t)h * N + irow];
        const float mi = mv[(size_t)h * N + irow];
        const float rz = 1.f / zv[(size_t)h * N + irow];
        const float ac = acv[h], bc = bcv[h];
        const float* __restrict__ sd   = sdst + (size_t)h * N;
        const __bf16* __restrict__ Hpt = HT + (size_t)h * (size_t)D2 * N;

        for (int jt = 0; jt < N; jt += 32) {
            __syncthreads();
            {   // stage Hpt[:, jt..jt+32): D2 rows x 64B = 1024 b128 chunks
#pragma unroll
                for (int c = 0; c < (D2 * 4) / 128; ++c) {
                    const int idx = c * 128 + tid;
                    const int d = idx >> 2, part = (idx & 3) * 8;
                    async_load_b128(ldsB + (unsigned)idx * 16,
                                    Hpt + (size_t)d * N + jt + part);
                }
            }
            wait_async();
            __syncthreads();

            float av[16], sv[16];
            load_frag_rows(av, arow + jt + half * 8);
            load_frag_rows(sv, sd + jt + half * 8);
            v16bf P;
#pragma unroll
            for (int e = 0; e < 16; ++e) {
                const float aij = av[e];
                const float eij = si + sv[e];
                const float t   = fmaf(ac, aij, bc);
                const float lk  = (t >= 0.f) ? t : 0.2f * t;
                const float att = (aij > 0.f) ? lk * eij : MASK_VAL;
                P[e] = (__bf16)(__expf(att - mi) * rz);
            }
#pragma unroll
            for (int t = 0; t < 16; ++t) {
                const v16bf B = load_b_frag(T32, t * 16 + n16, half);
                acc[t] = wmma_bf16(P, B, acc[t]);
            }
        }
    }
    const float inv_h = 1.f / (float)H2;
#pragma unroll
    for (int t = 0; t < 16; ++t) {
#pragma unroll
        for (int r = 0; r < 8; ++r) {
            const int M = r + 8 * half;
            const float v = acc[t][r] * inv_h;                 // mean over heads
            const float o = (v > 0.f) ? v : (__expf(v) - 1.f); // ELU
            OutBuf[(size_t)(i0 + M) * D2 + t * 16 + n16] = o;
        }
    }
}

// ---------------------------------------------------------------------------
__global__ void k_colmax(const float* __restrict__ OutBuf, float* __restrict__ outmax,
                         int N, int D) {
    const int d = threadIdx.x;
    float m = -3.4e38f;
    for (int n = 0; n < N; ++n) m = fmaxf(m, OutBuf[(size_t)n * D + d]);
    outmax[d] = m;
}

__global__ void k_final(const float* __restrict__ outmax, const float* __restrict__ Wm,
                        const float* __restrict__ bm, float* __restrict__ out, int D) {
    const int j = threadIdx.x;
    float s = bm[j];
    for (int d = 0; d < D; ++d) s = fmaf(outmax[d], Wm[(size_t)d * D + j], s);
    out[j] = s;
}

// ---------------------------------------------------------------------------
extern "C" void kernel_launch(void* const* d_in, const int* in_sizes, int n_in,
                              void* d_out, int out_size, void* d_ws, size_t ws_size,
                              hipStream_t stream) {
    (void)in_sizes; (void)n_in; (void)out_size; (void)ws_size;
    const float* x   = (const float*)d_in[0];
    const float* adj = (const float*)d_in[1];
    const float* w1  = (const float*)d_in[2];
    const float* a1  = (const float*)d_in[3];
    const float* ac1 = (const float*)d_in[4];
    const float* bc1 = (const float*)d_in[5];
    const float* w2  = (const float*)d_in[6];
    const float* a2  = (const float*)d_in[7];
    const float* ac2 = (const float*)d_in[8];
    const float* bc2 = (const float*)d_in[9];
    const float* Wm  = (const float*)d_in[10];
    const float* bm  = (const float*)d_in[11];
    float* out = (float*)d_out;

    // byte-addressed workspace carve-up (256B aligned slots)
    char* base = (char*)d_ws;
    size_t off = 0;
    auto carve = [&](size_t bytes) { char* p = base + off;
                                     off += (bytes + 255) & ~(size_t)255; return p; };
    __bf16* xbf   = (__bf16*)carve((size_t)N_NODES * F_IN * 2);
    __bf16* w1t   = (__bf16*)carve((size_t)H1 * F_IN * D1 * 2);
    __bf16* w2t   = (__bf16*)carve((size_t)H2 * FIN2 * D2 * 2);
    float*  h1    = (float*) carve((size_t)H1 * N_NODES * D1 * 4);
    __bf16* h1t   = (__bf16*)carve((size_t)H1 * D1 * N_NODES * 2);
    float*  ssrc1 = (float*) carve((size_t)H1 * N_NODES * 4);
    float*  sdst1 = (float*) carve((size_t)H1 * N_NODES * 4);
    float*  m1    = (float*) carve((size_t)H1 * N_NODES * 4);
    float*  z1    = (float*) carve((size_t)H1 * N_NODES * 4);
    __bf16* hcat  = (__bf16*)carve((size_t)N_NODES * FIN2 * 2);
    float*  h2    = (float*) carve((size_t)H2 * N_NODES * D2 * 4);
    __bf16* h2t   = (__bf16*)carve((size_t)H2 * D2 * N_NODES * 2);
    float*  ssrc2 = (float*) carve((size_t)H2 * N_NODES * 4);
    float*  sdst2 = (float*) carve((size_t)H2 * N_NODES * 4);
    float*  m2    = (float*) carve((size_t)H2 * N_NODES * 4);
    float*  z2    = (float*) carve((size_t)H2 * N_NODES * 4);
    float*  outbuf= (float*) carve((size_t)N_NODES * D2 * 4);
    float*  outmax= (float*) carve((size_t)D2 * 4);

    // ---- input conversions ----
    const int nx = N_NODES * F_IN;
    k_f32_to_bf16<<<(nx + 255) / 256, 256, 0, stream>>>(x, xbf, nx);
    const int nw1 = H1 * F_IN * D1;
    k_w_transpose<<<(nw1 + 255) / 256, 256, 0, stream>>>(w1, w1t, F_IN, D1, nw1);
    const int nw2 = H2 * FIN2 * D2;
    k_w_transpose<<<(nw2 + 255) / 256, 256, 0, stream>>>(w2, w2t, FIN2, D2, nw2);

    // ---- layer 1 ----
    k_proj<F_IN, D1><<<dim3(N_NODES / 64, D1 / 16, H1), 128, 0, stream>>>(
        xbf, w1t, h1, h1t, N_NODES);
    k_scores<<<dim3(N_NODES / 256, H1), 256, 0, stream>>>(h1, a1, ssrc1, sdst1,
                                                          N_NODES, D1);
    k_stats<<<dim3(N_NODES / 256, H1), 256, 0, stream>>>(adj, ssrc1, sdst1, ac1, bc1,
                                                         m1, z1, N_NODES);
    k_attn1<<<dim3(N_NODES / 64, H1), 128, 0, stream>>>(adj, h1t, ssrc1, sdst1,
                                                        m1, z1, ac1, bc1, hcat, N_NODES);
    // ---- layer 2 ----
    k_proj<FIN2, D2><<<dim3(N_NODES / 64, D2 / 16, H2), 128, 0, stream>>>(
        hcat, w2t, h2, h2t, N_NODES);
    k_scores<<<dim3(N_NODES / 256, H2), 256, 0, stream>>>(h2, a2, ssrc2, sdst2,
                                                          N_NODES, D2);
    k_stats<<<dim3(N_NODES / 256, H2), 256, 0, stream>>>(adj, ssrc2, sdst2, ac2, bc2,
                                                         m2, z2, N_NODES);
    k_attn2<<<dim3(N_NODES / 64, 1), 128, 0, stream>>>(adj, h2t, ssrc2, sdst2,
                                                       m2, z2, ac2, bc2, outbuf, N_NODES);
    // ---- readout ----
    k_colmax<<<1, D2, 0, stream>>>(outbuf, outmax, N_NODES, D2);
    k_final<<<1, D2, 0, stream>>>(outmax, Wm, bm, out, D2);
}
